// MoELayer_84679575208533
// MI455X (gfx1250) — matrix operand order repaired and checked by reading, not compiled, not run on hardware
//
#include <hip/hip_runtime.h>
#include <hip/hip_bf16.h>
#include <math.h>

// ---------------- problem constants ----------------
#define B_SZ   4
#define T_SZ   2048
#define D_DIM  2048
#define E_NUM  8
#define I_DIM  1024
#define N_TOK  (B_SZ * T_SZ)     // 8192 tokens
#define ND_OUT ((size_t)N_TOK * D_DIM)

// tiling
#define KCH 32                    // K chunk per WMMA step
#define TM  64                    // token rows per block
#define TN  64                    // output cols per block
#define AS  40                    // LDS row stride (f16) = 32 + 8 pad

typedef __attribute__((ext_vector_type(16))) _Float16 v16h;
typedef __attribute__((ext_vector_type(8)))  _Float16 v8h;
typedef __attribute__((ext_vector_type(8)))  float    v8f;
typedef __attribute__((ext_vector_type(4)))  int      v4i;

union F16x16 { v16h v; v8h h[2]; };

__device__ __forceinline__ v8f wmma_f16(v16h a, v16h b, v8f c) {
  return __builtin_amdgcn_wmma_f32_16x16x32_f16(false, a, false, b,
                                                (short)0, c, false, false);
}

// fast 1/x via v_rcp_f32 (result is truncated to f16 anyway)
__device__ __forceinline__ float fast_rcp(float x) {
#if defined(__has_builtin)
#  if __has_builtin(__builtin_amdgcn_rcpf)
  return __builtin_amdgcn_rcpf(x);
#  else
  return 1.0f / x;
#  endif
#else
  return 1.0f / x;
#endif
}

// ---------------- async copy global->LDS (CDNA5), with sync fallback ----------
#if defined(__has_builtin)
#  if __has_builtin(__builtin_amdgcn_global_load_async_to_lds_b128) && \
      __has_builtin(__builtin_amdgcn_s_wait_asynccnt)
#    define MOE_ASYNC 1
#  endif
#endif
#ifndef MOE_ASYNC
#  define MOE_ASYNC 0
#endif

__device__ __forceinline__ void cp16(void* l, const void* g) {
#if MOE_ASYNC
  __builtin_amdgcn_global_load_async_to_lds_b128(
      (__attribute__((address_space(1))) v4i*)(void*)g,
      (__attribute__((address_space(3))) v4i*)l, 0, 0);
#else
  *(v8h*)l = *(const v8h*)g;
#endif
}

__device__ __forceinline__ void cp_wait() {
#if MOE_ASYNC
  __builtin_amdgcn_s_wait_asynccnt(0);
#endif
}

// ---------------- kernel 0: zero output + ws header ----------------
__global__ void moe_zero_kernel(float* __restrict__ out, int n, int* __restrict__ hdr) {
  int i = blockIdx.x * 256 + threadIdx.x;
  if (i < n) out[i] = 0.0f;
  if (i < 16) hdr[i] = 0;           // counts[8] + psum[8]
}

// ---------------- kernel 0b: fp32 -> f16 bulk convert ----------------
__global__ void __launch_bounds__(256)
moe_cvt_kernel(const float* __restrict__ src, _Float16* __restrict__ dst, int n8) {
  int i = blockIdx.x * 256 + threadIdx.x;
  if (i >= n8) return;
  const float4* p = (const float4*)(src + (size_t)i * 8);
  float4 a = p[0], b = p[1];
  v8h o;
  o[0] = (_Float16)a.x; o[1] = (_Float16)a.y;
  o[2] = (_Float16)a.z; o[3] = (_Float16)a.w;
  o[4] = (_Float16)b.x; o[5] = (_Float16)b.y;
  o[6] = (_Float16)b.z; o[7] = (_Float16)b.w;
  *(v8h*)(dst + (size_t)i * 8) = o;
}

// ---------------- kernel 1: gating / routing ----------------
__global__ void __launch_bounds__(64)
moe_gate_kernel(const float* __restrict__ x, const float* __restrict__ gw,
                int* __restrict__ counts, float* __restrict__ psum,
                int* __restrict__ tlist, float* __restrict__ tw) {
  __shared__ __align__(16) float gs[E_NUM][128];
  const int tid = threadIdx.x;
  const int tok = blockIdx.x * 64 + tid;

  float acc[E_NUM];
#pragma unroll
  for (int e = 0; e < E_NUM; ++e) acc[e] = 0.0f;

  for (int k0 = 0; k0 < D_DIM; k0 += 128) {
#pragma unroll
    for (int q = 0; q < 4; ++q) {
      int f = tid * 16 + q * 4;
      int r = f >> 7, c = f & 127;
      *(float4*)&gs[r][c] = *(const float4*)(gw + (size_t)r * D_DIM + k0 + c);
    }
    __syncthreads();
    const float* xr = x + (size_t)tok * D_DIM + k0;
    for (int kk = 0; kk < 128; kk += 4) {
      float4 xv = *(const float4*)(xr + kk);
#pragma unroll
      for (int e = 0; e < E_NUM; ++e) {
        acc[e] += xv.x * gs[e][kk]     + xv.y * gs[e][kk + 1] +
                  xv.z * gs[e][kk + 2] + xv.w * gs[e][kk + 3];
      }
    }
    __syncthreads();
  }

  float m = acc[0];
#pragma unroll
  for (int e = 1; e < E_NUM; ++e) m = fmaxf(m, acc[e]);
  float p[E_NUM], s = 0.0f;
#pragma unroll
  for (int e = 0; e < E_NUM; ++e) { p[e] = __expf(acc[e] - m); s += p[e]; }
  float inv = 1.0f / s;
#pragma unroll
  for (int e = 0; e < E_NUM; ++e) {
    p[e] *= inv;
    atomicAdd(&psum[e], p[e]);
  }
  int i1 = 0;
#pragma unroll
  for (int e = 1; e < E_NUM; ++e) if (p[e] > p[i1]) i1 = e;
  int i2 = (i1 == 0) ? 1 : 0;
#pragma unroll
  for (int e = 0; e < E_NUM; ++e) if (e != i1 && p[e] > p[i2]) i2 = e;
  float sw = 1.0f / (p[i1] + p[i2]);

  int pos1 = atomicAdd(&counts[i1], 1);
  tlist[i1 * N_TOK + pos1] = tok;
  tw[i1 * N_TOK + pos1] = p[i1] * sw;
  int pos2 = atomicAdd(&counts[i2], 1);
  tlist[i2 * N_TOK + pos2] = tok;
  tw[i2 * N_TOK + pos2] = p[i2] * sw;
}

// ---------------- kernel 2: H = silu(X w1^T) * (X w3^T) ----------------
// 128 threads = 4 waves. Tile TM=64 rows x TN=64 cols; wave wv owns n-subtile,
// loops 4 m-subtiles -> 8 WMMA per K chunk.
__global__ void __launch_bounds__(128)
moe_pass1_kernel(const _Float16* __restrict__ xh,
                 const _Float16* __restrict__ w1h, const _Float16* __restrict__ w3h,
                 const int* __restrict__ counts, const int* __restrict__ tlist,
                 _Float16* __restrict__ Hbuf) {
  const int e = blockIdx.z;
  const int cnt = counts[e];
  const int row0 = blockIdx.y * TM;
  if (row0 >= cnt) return;
  const int i0 = blockIdx.x * TN;

  int rs = 0;
  for (int j = 0; j < e; ++j) rs += counts[j];

  __shared__ __align__(16) _Float16 At[TM][AS];
  __shared__ __align__(16) _Float16 W1t[TN][AS];
  __shared__ __align__(16) _Float16 W3t[TN][AS];
  __shared__ int toks[TM];

  const int tid = threadIdx.x;
  if (tid < TM) {
    int rr = row0 + tid;
    if (rr >= cnt) rr = cnt - 1;          // clamp pads; stores are guarded
    toks[tid] = tlist[e * N_TOK + rr];
  }
  __syncthreads();

  const int lane = tid & 31, wv = tid >> 5;
  const int lr = lane & 15, hi = lane >> 4;

  v8f acc1[4] = {{0,0,0,0,0,0,0,0},{0,0,0,0,0,0,0,0},
                 {0,0,0,0,0,0,0,0},{0,0,0,0,0,0,0,0}};
  v8f acc3[4] = {{0,0,0,0,0,0,0,0},{0,0,0,0,0,0,0,0},
                 {0,0,0,0,0,0,0,0},{0,0,0,0,0,0,0,0}};

  for (int k0 = 0; k0 < D_DIM; k0 += KCH) {
#pragma unroll
    for (int q = 0; q < 2; ++q) {        // A tile: 64x32 f16 = 256 16B chunks
      int ch = tid * 2 + q;
      int r = ch >> 2, c = (ch & 3) * 8;
      cp16(&At[r][c], xh + (size_t)toks[r] * D_DIM + k0 + c);
    }
#pragma unroll
    for (int q = 0; q < 2; ++q) {        // W1/W3 tiles: 64x32 f16 each
      int ch = tid * 2 + q;
      int r = ch >> 2, c = (ch & 3) * 8;
      size_t wo = ((size_t)e * I_DIM + i0 + r) * D_DIM + k0 + c;
      cp16(&W1t[r][c], w1h + wo);
      cp16(&W3t[r][c], w3h + wo);
    }
    cp_wait();
    __syncthreads();

    F16x16 b1, b3;
    const int wr = wv * 16 + lr;
    b1.h[0] = *(const v8h*)&W1t[wr][hi * 8];
    b1.h[1] = *(const v8h*)&W1t[wr][16 + hi * 8];
    b3.h[0] = *(const v8h*)&W3t[wr][hi * 8];
    b3.h[1] = *(const v8h*)&W3t[wr][16 + hi * 8];
#pragma unroll
    for (int mm = 0; mm < 4; ++mm) {
      F16x16 a;
      a.h[0] = *(const v8h*)&At[mm * 16 + lr][hi * 8];
      a.h[1] = *(const v8h*)&At[mm * 16 + lr][16 + hi * 8];
      acc1[mm] = wmma_f16(a.v, b1.v, acc1[mm]);
      acc3[mm] = wmma_f16(a.v, b3.v, acc3[mm]);
    }
    __syncthreads();
  }

  const int col = i0 + wv * 16 + lr;
#pragma unroll
  for (int mm = 0; mm < 4; ++mm) {
#pragma unroll
    for (int j = 0; j < 8; ++j) {
      int rl = mm * 16 + j + hi * 8;
      if (row0 + rl < cnt) {
        float g = acc1[mm][j];
        // silu(g) = g * sigmoid(g); v_rcp_f32 is plenty for f16 output
        float hval = g * fast_rcp(1.0f + __expf(-g)) * acc3[mm][j];
        Hbuf[(size_t)(rs + row0 + rl) * I_DIM + col] = (_Float16)hval;
      }
    }
  }
}

// ---------------- kernel 3: out += w_tok * (H w2^T) ----------------
__global__ void __launch_bounds__(128)
moe_pass2_kernel(const _Float16* __restrict__ Hbuf, const _Float16* __restrict__ w2h,
                 const int* __restrict__ counts, const int* __restrict__ tlist,
                 const float* __restrict__ tw, float* __restrict__ out) {
  const int e = blockIdx.z;
  const int cnt = counts[e];
  const int row0 = blockIdx.y * TM;
  if (row0 >= cnt) return;
  const int d0 = blockIdx.x * TN;

  int rs = 0;
  for (int j = 0; j < e; ++j) rs += counts[j];

  __shared__ __align__(16) _Float16 At[TM][AS];
  __shared__ __align__(16) _Float16 Wt[TN][AS];
  __shared__ int   toks[TM];
  __shared__ float wgts[TM];

  const int tid = threadIdx.x;
  if (tid < TM) {
    int rr = row0 + tid;
    if (rr >= cnt) rr = cnt - 1;
    toks[tid] = tlist[e * N_TOK + rr];
    wgts[tid] = tw[e * N_TOK + rr];
  }
  __syncthreads();

  const int lane = tid & 31, wv = tid >> 5;
  const int lr = lane & 15, hi = lane >> 4;

  v8f acc[4] = {{0,0,0,0,0,0,0,0},{0,0,0,0,0,0,0,0},
                {0,0,0,0,0,0,0,0},{0,0,0,0,0,0,0,0}};

  for (int k0 = 0; k0 < I_DIM; k0 += KCH) {
#pragma unroll
    for (int q = 0; q < 2; ++q) {        // A tile from H rows (clamped)
      int ch = tid * 2 + q;
      int r = ch >> 2, c = (ch & 3) * 8;
      int rr = row0 + r; if (rr >= cnt) rr = cnt - 1;
      cp16(&At[r][c], Hbuf + (size_t)(rs + rr) * I_DIM + k0 + c);
    }
#pragma unroll
    for (int q = 0; q < 2; ++q) {        // w2 tile: 64 d-rows x 32 i
      int ch = tid * 2 + q;
      int r = ch >> 2, c = (ch & 3) * 8;
      cp16(&Wt[r][c], w2h + ((size_t)e * D_DIM + d0 + r) * I_DIM + k0 + c);
    }
    cp_wait();
    __syncthreads();

    F16x16 b;
    const int wr = wv * 16 + lr;
    b.h[0] = *(const v8h*)&Wt[wr][hi * 8];
    b.h[1] = *(const v8h*)&Wt[wr][16 + hi * 8];
#pragma unroll
    for (int mm = 0; mm < 4; ++mm) {
      F16x16 a;
      a.h[0] = *(const v8h*)&At[mm * 16 + lr][hi * 8];
      a.h[1] = *(const v8h*)&At[mm * 16 + lr][16 + hi * 8];
      acc[mm] = wmma_f16(a.v, b.v, acc[mm]);
    }
    __syncthreads();
  }

  const int col = d0 + wv * 16 + lr;
#pragma unroll
  for (int mm = 0; mm < 4; ++mm) {
#pragma unroll
    for (int j = 0; j < 8; ++j) {
      int rl = mm * 16 + j + hi * 8;
      if (row0 + rl < cnt) {
        atomicAdd(&out[(size_t)toks[rl] * D_DIM + col], wgts[rl] * acc[mm][j]);
      }
    }
  }
}

// ---------------- kernel 4: aux loss ----------------
__global__ void moe_aux_kernel(const int* __restrict__ counts,
                               const float* __restrict__ psum,
                               float* __restrict__ out) {
  if (threadIdx.x == 0 && blockIdx.x == 0) {
    float a = 0.0f;
    for (int e = 0; e < E_NUM; ++e) {
      float f = (float)counts[e] / (float)N_TOK;
      float p = psum[e] / (float)N_TOK;
      a += f * p;
    }
    out[ND_OUT] = 0.01f * (float)E_NUM * a;
  }
}

// ---------------- launch ----------------
extern "C" void kernel_launch(void* const* d_in, const int* in_sizes, int n_in,
                              void* d_out, int out_size, void* d_ws, size_t ws_size,
                              hipStream_t stream) {
  (void)in_sizes; (void)n_in; (void)ws_size;
  const float* x  = (const float*)d_in[0];
  const float* gw = (const float*)d_in[1];
  const float* w1 = (const float*)d_in[2];
  const float* w3 = (const float*)d_in[3];
  const float* w2 = (const float*)d_in[4];
  float* out = (float*)d_out;

  // workspace layout (bytes)
  const size_t SZ_LIST = (size_t)E_NUM * N_TOK * 4;      // 256 KB
  const size_t SZ_H    = (size_t)2 * N_TOK * I_DIM * 2;  // 32 MB
  const size_t SZ_XH   = (size_t)N_TOK * D_DIM * 2;      // 32 MB
  const size_t SZ_W    = (size_t)E_NUM * I_DIM * D_DIM * 2; // 32 MB each

  char* ws = (char*)d_ws;
  int*      counts = (int*)ws;                      // [8]
  float*    psum   = (float*)(ws + 32);             // [8]
  int*      tlist  = (int*)(ws + 256);
  float*    tw     = (float*)(ws + 256 + SZ_LIST);
  _Float16* H      = (_Float16*)(ws + 524800);
  _Float16* xh     = (_Float16*)(ws + 524800 + SZ_H);
  _Float16* w1h    = (_Float16*)(ws + 524800 + SZ_H + SZ_XH);
  _Float16* w3h    = (_Float16*)(ws + 524800 + SZ_H + SZ_XH + SZ_W);
  _Float16* w2h    = (_Float16*)(ws + 524800 + SZ_H + SZ_XH + 2 * SZ_W);

  // 1) zero output + counters
  {
    int n = out_size;
    moe_zero_kernel<<<(n + 255) / 256, 256, 0, stream>>>(out, n, counts);
  }
  // 2) fp32 -> f16 pre-conversion (x, w1, w3, w2)
  {
    int n8 = (int)(SZ_XH / 16);   // elements/8
    moe_cvt_kernel<<<(n8 + 255) / 256, 256, 0, stream>>>(x,  xh,  n8);
    int w8 = (int)(SZ_W / 16);
    moe_cvt_kernel<<<(w8 + 255) / 256, 256, 0, stream>>>(w1, w1h, w8);
    moe_cvt_kernel<<<(w8 + 255) / 256, 256, 0, stream>>>(w3, w3h, w8);
    moe_cvt_kernel<<<(w8 + 255) / 256, 256, 0, stream>>>(w2, w2h, w8);
  }
  // 3) gate / route
  moe_gate_kernel<<<N_TOK / 64, 64, 0, stream>>>(x, gw, counts, psum, tlist, tw);
  // 4) H = silu(X w1^T) * (X w3^T)
  {
    dim3 grid(I_DIM / TN, N_TOK / TM, E_NUM);
    moe_pass1_kernel<<<grid, 128, 0, stream>>>(xh, w1h, w3h, counts, tlist, H);
  }
  // 5) out += w * (H w2^T)
  {
    dim3 grid(D_DIM / TN, N_TOK / TM, E_NUM);
    moe_pass2_kernel<<<grid, 128, 0, stream>>>(H, w2h, counts, tlist, tw, out);
  }
  // 6) aux loss scalar
  moe_aux_kernel<<<1, 32, 0, stream>>>(counts, psum, out);
}
